// Sequence_9509057593523
// MI455X (gfx1250) — compile-verified, matrix operand order checked by reading, not run on hardware
//
#include <hip/hip_runtime.h>
#include <hip/hip_bf16.h>

// ---------------------------------------------------------------------------
// Persistent 2-layer LSTM for MI455X (gfx1250, wave32, WMMA).
//
//   - Weights R1, W2, R2 pre-converted to f16 and stored in LDS in the exact
//     per-lane VGPR layout of the WMMA B operand (32x16 f16, K-major blocks),
//     so a B operand is two contiguous ds_read_b128 per lane.
//   - h state is staged in LDS as row-major [16 x 64] f16 so the next step's
//     A operand (16x32 f16) is two contiguous 16B reads per lane.
//   - Gate GEMM columns laid out as (coltile, gate) interleaved: 16 N-tiles,
//     wave w owns coltile w -> i/f/g/o tiles for the same 16 hidden units are
//     computed by one wave and combined in registers (c state never leaves
//     VGPRs).
//   - Math: v_wmma_f32_16x16x32_f16 (f32 accumulate); activations are
//     branch-free v_exp2/v_rcp sequences (TRANS ops co-execute with WMMA).
// ---------------------------------------------------------------------------

#define HH   51
#define HP   64          // padded hidden (K padded to 2x32)
#define LSEQ 1000
#define NB   1024

#define ROWS_PER_BLOCK 16
#define WAVES          4
#define NBLOCKS        (NB / ROWS_PER_BLOCK)   // 64

#define BLK_HALVES 512                    // one 32x16 f16 B block
#define MAT_HALVES (16 * 2 * BLK_HALVES)  // 16 N-tiles x 2 K-blocks = 16384

typedef __attribute__((ext_vector_type(16))) _Float16 v16h;
typedef __attribute__((ext_vector_type(8)))  _Float16 v8h;
typedef __attribute__((ext_vector_type(8)))  float    v8f;

#define LOG2E 1.4426950408889634f

// Branch-free sigmoid / tanh on the hardware transcendental pipe:
//   sigm(x) = rcp(1 + exp2(-log2e * x))
//   tanh(x) = 2*sigm(2x) - 1
__device__ __forceinline__ float sigm(float x) {
    float e = __builtin_amdgcn_exp2f(x * (-LOG2E));
    return __builtin_amdgcn_rcpf(1.0f + e);
}
__device__ __forceinline__ float tanh_fast(float x) {
    float e = __builtin_amdgcn_exp2f(x * (-2.0f * LOG2E));
    return __builtin_fmaf(2.0f, __builtin_amdgcn_rcpf(1.0f + e), -1.0f);
}

__device__ __forceinline__ v8f vzero8() {
    v8f z;
#pragma unroll
    for (int r = 0; r < 8; ++r) z[r] = 0.0f;
    return z;
}

// A operand (16x32 f16, M = lane&15, K-half = lane>>4) from row-major [16][64]
// f16 staging. ISA layout: lane group g holds K = 8g+0..7 and K = 16+8g+0..7
// (plus kb*32).
__device__ __forceinline__ v16h load_A(const _Float16* stage, int lcol, int lhalf, int kb) {
    const v8h* row = (const v8h*)(stage + lcol * HP);
    v8h lo = row[kb * 4 + lhalf];
    v8h hi = row[kb * 4 + 2 + lhalf];
    v16h a;
#pragma unroll
    for (int e = 0; e < 8; ++e) { a[e] = lo[e]; a[8 + e] = hi[e]; }
    return a;
}

// B operand: stored pre-swizzled, lane's 16 halves contiguous at lane*16.
__device__ __forceinline__ v16h load_B(const _Float16* blk, int lane) {
    const v8h* q = (const v8h*)blk + lane * 2;
    v8h lo = q[0];
    v8h hi = q[1];
    v16h b;
#pragma unroll
    for (int e = 0; e < 8; ++e) { b[e] = lo[e]; b[8 + e] = hi[e]; }
    return b;
}

__global__ __launch_bounds__(WAVES * 32, 1)
void lstm2_persistent_kernel(const float* __restrict__ input,
                             const float* __restrict__ W1,
                             const float* __restrict__ b1,
                             const float* __restrict__ R1,
                             const float* __restrict__ W2,
                             const float* __restrict__ b2,
                             const float* __restrict__ R2,
                             const float* __restrict__ w_out,
                             const float* __restrict__ b_out,
                             float* __restrict__ out)
{
    // --- LDS ---------------------------------------------------------------
    __shared__ __attribute__((aligned(16))) _Float16 sB[3 * MAT_HALVES]; // R1|W2|R2, 96 KB
    __shared__ __attribute__((aligned(16))) float    sBias[2 * 4 * HP];  // b1p | b2p
    __shared__ __attribute__((aligned(16))) float    sW1c[4 * HP];       // W1[:, :, 0] padded
    __shared__ __attribute__((aligned(16))) float    sWo[HP];            // w_out padded
    __shared__ __attribute__((aligned(16))) _Float16 sH0[ROWS_PER_BLOCK * HP]; // h1 staging
    __shared__ __attribute__((aligned(16))) _Float16 sH1[ROWS_PER_BLOCK * HP]; // h2 staging

    const int tid   = threadIdx.x;
    const int lane  = tid & 31;
    const int wave  = tid >> 5;       // wave == coltile it owns (0..3)
    const int lcol  = lane & 15;
    const int lhalf = lane >> 4;
    const int ct    = wave;

    // --- Build B matrices in WMMA B-operand order --------------------------
    // N-tile nt: gate = nt&3, coltile = nt>>2.  B[k_in][n] = R[gate][col][k_in]
    // (einsum 'nh,kgh->nkg').  Zero-pad col>=51 and k_in>=51.
    const float* srcs[3] = { R1, W2, R2 };
#pragma unroll
    for (int m = 0; m < 3; ++m) {
        const float* src = srcs[m];
        for (int eid = tid; eid < MAT_HALVES; eid += blockDim.x) {
            int blk    = eid >> 9;        // nt*2 + kb
            int within = eid & 511;
            int bl     = within >> 4;     // destination lane
            int e      = within & 15;     // element within lane's v16h
            int nt = blk >> 1, kb = blk & 1;
            int bg = bl >> 4;
            int K  = (e < 8) ? (8 * bg + e) : (16 + 8 * bg + (e - 8));
            int k_in = kb * 32 + K;
            int gate = nt & 3;
            int col  = (nt >> 2) * 16 + (bl & 15);
            float v = (col < HH && k_in < HH) ? src[(gate * HH + col) * HH + k_in] : 0.0f;
            sB[m * MAT_HALVES + eid] = (_Float16)v;
        }
    }
    for (int i = tid; i < 4 * HP; i += blockDim.x) {
        int gate = i >> 6, col = i & 63;
        sBias[i]          = (col < HH) ? b1[gate * HH + col] : 0.0f;
        sBias[4 * HP + i] = (col < HH) ? b2[gate * HH + col] : 0.0f;
        sW1c[i]           = (col < HH) ? W1[gate * HH + col] : 0.0f;   // IN == 1
    }
    for (int i = tid; i < HP; i += blockDim.x) sWo[i] = (i < HH) ? w_out[i] : 0.0f;
    for (int i = tid; i < ROWS_PER_BLOCK * HP; i += blockDim.x) {
        sH0[i] = (_Float16)0.0f;
        sH1[i] = (_Float16)0.0f;
    }
    __syncthreads();

    const int   rowbase = blockIdx.x * ROWS_PER_BLOCK;
    const float bout    = b_out[0];

    // Cell state for this wave's coltile, C-matrix layout:
    // element r <-> row m = r + 8*lhalf, column ct*16 + lcol.
    v8f c1s = vzero8();
    v8f c2s = vzero8();
    const int colg = ct * 16 + lcol;

    for (int t = 0; t < LSEQ; ++t) {
        // x_t broadcast: lane holds x for batch row lcol; gather the 8 rows
        // this lane's accumulator elements correspond to.
        float xv = input[(rowbase + lcol) * LSEQ + t];
        float xm[8];
#pragma unroll
        for (int r = 0; r < 8; ++r) xm[r] = __shfl(xv, r + 8 * lhalf, 32);

        // ================= cell 1: g = x*W1 + h1*R1 + b1 =================
        v16h a1_0 = load_A(sH0, lcol, lhalf, 0);
        v16h a1_1 = load_A(sH0, lcol, lhalf, 1);

        v8f acc[4];
#pragma unroll
        for (int gt = 0; gt < 4; ++gt) {
            const int nt = ct * 4 + gt;
            const float bia = sBias[gt * HP + colg];
            const float wcl = sW1c[gt * HP + colg];
            v8f g;
#pragma unroll
            for (int r = 0; r < 8; ++r) g[r] = bia + xm[r] * wcl;
            g = __builtin_amdgcn_wmma_f32_16x16x32_f16(
                    false, a1_0, false, load_B(sB + (nt * 2 + 0) * BLK_HALVES, lane),
                    (short)0, g, false, false);
            g = __builtin_amdgcn_wmma_f32_16x16x32_f16(
                    false, a1_1, false, load_B(sB + (nt * 2 + 1) * BLK_HALVES, lane),
                    (short)0, g, false, false);
            acc[gt] = g;
        }
#pragma unroll
        for (int r = 0; r < 8; ++r) {
            float ig = sigm(acc[0][r]);
            float fg = sigm(acc[1][r]);
            float cd = tanh_fast(acc[2][r]);
            float og = sigm(acc[3][r]);
            float cn = __builtin_fmaf(ig, cd, fg * c1s[r]);
            c1s[r]   = cn;
            float hn = og * tanh_fast(cn);
            sH0[(r + 8 * lhalf) * HP + colg] = (_Float16)hn;
        }
        __syncthreads();   // h1 transpose visible to all waves

        // ============ cell 2: g = h1*W2 + h2*R2 + b2 =====================
        v16h a2_0 = load_A(sH0, lcol, lhalf, 0);
        v16h a2_1 = load_A(sH0, lcol, lhalf, 1);
        v16h ah_0 = load_A(sH1, lcol, lhalf, 0);
        v16h ah_1 = load_A(sH1, lcol, lhalf, 1);

#pragma unroll
        for (int gt = 0; gt < 4; ++gt) {
            const int nt = ct * 4 + gt;
            const float bia = sBias[4 * HP + gt * HP + colg];
            v8f g;
#pragma unroll
            for (int r = 0; r < 8; ++r) g[r] = bia;
            g = __builtin_amdgcn_wmma_f32_16x16x32_f16(
                    false, a2_0, false, load_B(sB + MAT_HALVES + (nt * 2 + 0) * BLK_HALVES, lane),
                    (short)0, g, false, false);
            g = __builtin_amdgcn_wmma_f32_16x16x32_f16(
                    false, a2_1, false, load_B(sB + MAT_HALVES + (nt * 2 + 1) * BLK_HALVES, lane),
                    (short)0, g, false, false);
            g = __builtin_amdgcn_wmma_f32_16x16x32_f16(
                    false, ah_0, false, load_B(sB + 2 * MAT_HALVES + (nt * 2 + 0) * BLK_HALVES, lane),
                    (short)0, g, false, false);
            g = __builtin_amdgcn_wmma_f32_16x16x32_f16(
                    false, ah_1, false, load_B(sB + 2 * MAT_HALVES + (nt * 2 + 1) * BLK_HALVES, lane),
                    (short)0, g, false, false);
            acc[gt] = g;
        }
#pragma unroll
        for (int r = 0; r < 8; ++r) {
            float ig = sigm(acc[0][r]);
            float fg = sigm(acc[1][r]);
            float cd = tanh_fast(acc[2][r]);
            float og = sigm(acc[3][r]);
            float cn = __builtin_fmaf(ig, cd, fg * c2s[r]);
            c2s[r]   = cn;
            float hn = og * tanh_fast(cn);
            sH1[(r + 8 * lhalf) * HP + colg] = (_Float16)hn;
        }
        __syncthreads();   // full h2 row visible for the output projection

        // ============ out[:, t] = h2 . w_out + b_out =====================
        if (wave == 0 && lane < 16) {
            const _Float16* hrow = sH1 + lane * HP;
            float a = bout;
#pragma unroll
            for (int j = 0; j < HP; ++j) a = __builtin_fmaf((float)hrow[j], sWo[j], a);
            out[(rowbase + lane) * LSEQ + t] = a;
        }
        __syncthreads();   // protect sH1 against next step's cell-2 writes
    }
}

extern "C" void kernel_launch(void* const* d_in, const int* in_sizes, int n_in,
                              void* d_out, int out_size, void* d_ws, size_t ws_size,
                              hipStream_t stream) {
    (void)in_sizes; (void)n_in; (void)out_size; (void)d_ws; (void)ws_size;
    const float* input = (const float*)d_in[0];
    const float* W1    = (const float*)d_in[1];
    const float* b1    = (const float*)d_in[2];
    const float* R1    = (const float*)d_in[3];
    const float* W2    = (const float*)d_in[4];
    const float* b2    = (const float*)d_in[5];
    const float* R2    = (const float*)d_in[6];
    const float* w_out = (const float*)d_in[7];
    const float* b_out = (const float*)d_in[8];
    // d_in[9] == future (0): ignored.
    lstm2_persistent_kernel<<<NBLOCKS, WAVES * 32, 0, stream>>>(
        input, W1, b1, R1, W2, b2, R2, w_out, b_out, (float*)d_out);
}
